// PHGNS_NDAE_29214367547495
// MI455X (gfx1250) — matrix operand order repaired and checked by reading, not compiled, not run on hardware
//
#include <hip/hip_runtime.h>
#include <hip/hip_bf16.h>
#include <math.h>
#include <stdint.h>

// ---------------- dims ----------------
#define NCd 1024
#define NLd 1024
#define STATE_DIM 6656          // NC+NL+NN+NV
#define N_EDGES 4096
#define N_GNODES 4097
#define N_GPAD   4160           // padded to multiple of 64
#define N_CTRL   1024
#define NDIFF    2048
#define NALG     4608
#define DT_C     0.01f

typedef __attribute__((ext_vector_type(16))) _Float16 v16h;
typedef __attribute__((ext_vector_type(8)))  _Float16 v8h;
typedef __attribute__((ext_vector_type(4)))  _Float16 h4;
typedef __attribute__((ext_vector_type(8)))  float    v8f;
typedef __attribute__((ext_vector_type(4)))  float    f4;
typedef __attribute__((ext_vector_type(4)))  unsigned int u32x4;
typedef __attribute__((ext_vector_type(8)))  int      i32x8;
typedef __attribute__((ext_vector_type(4)))  int      i32x4;

#if defined(__has_builtin)
#  if __has_builtin(__builtin_amdgcn_tensor_load_to_lds) && __has_builtin(__builtin_amdgcn_s_wait_tensorcnt)
#    define HAVE_TDM 1
#  endif
#endif

__device__ __forceinline__ int edge_type(int row) {
    return (row >= 3584) ? 4 : (row >= 3072) ? 3 : (row >= 2048) ? 2 : (row >= 1024) ? 1 : 0;
}

// ============ weight pre-transpose: W[K x 128] f32 -> Wt[c*K + k] f16 ============
__global__ void w_transpose(const float* __restrict__ W, _Float16* __restrict__ Wt, int K)
{
    int idx = blockIdx.x * 256 + threadIdx.x;      // K*128 total
    int c = idx & 127, k = idx >> 7;
    Wt[(size_t)c * K + k] = (_Float16)W[(size_t)k * 128 + c];
}

// ============ WMMA GEMM: C[M x 128] = op(A[M x K] @ W[K x 128] + bias) ============
// REQUIRES: M (grid.x*64) multiple of 64, K multiple of 32, 16B alignment.
// Wt = f16 column-major transposed weights (col stride K).
// flags: 1 = ReLU, 2 = add residual (resid may alias C)
// block = 256 threads (8 waves): 64-row x 128-col tile; each wave 4 M-tiles x 16 cols.
// W tile (128 cols x 32 k) is DMA'd into LDS by the Tensor Data Mover when available.
__global__ __launch_bounds__(256)
void gemm128_wmma(const float* __restrict__ A, int lda,
                  const _Float16* __restrict__ Wt,      // transposed f16 weights
                  const float* __restrict__ bias,       // 128
                  const float* __restrict__ resid,      // M x 128 (or null)
                  float* __restrict__ C,                // M x 128
                  int K, int flags)
{
    __shared__ _Float16 lA[64][32];      // A tile, row-major f16
    __shared__ _Float16 lWt[128 * 32];   // W tile: col c at [c*32, c*32+32)
    const int tid  = threadIdx.x;
    const int lane = tid & 31;
    const int wave = tid >> 5;
    const int g    = lane >> 4;
    const int mr   = lane & 15;
    const int ncol = wave * 16 + mr;
    const int row0 = blockIdx.x * 64;

    v8f acc[4] = {};
    for (int k0 = 0; k0 < K; k0 += 32) {
        if (k0 + 32 < K)                  // prefetch next A tile (global_prefetch_b8)
            __builtin_prefetch(&A[(size_t)(row0 + (tid >> 2)) * lda + k0 + 32 + (tid & 3) * 8], 0, 1);

        // stage A tile: 64x32 f32 = 512 float4; cvt to f16, vector LDS store
#pragma unroll
        for (int it = 0; it < 2; ++it) {
            int idx = tid + it * 256;
            int r = idx >> 3, c4 = (idx & 7) * 4;
            f4 v = *(const f4*)&A[(size_t)(row0 + r) * lda + k0 + c4];
            h4 h; h[0] = (_Float16)v[0]; h[1] = (_Float16)v[1];
                  h[2] = (_Float16)v[2]; h[3] = (_Float16)v[3];
            *(h4*)&lA[r][c4] = h;
        }

#ifdef HAVE_TDM
        // Tensor Data Mover: one 2D descriptor moves the whole 8KB W tile.
        // tile_dim0 = 32 halves (k), tile_dim1 = 128 rows (cols), dim0 stride = K.
        if (wave == 0) {
            uint64_t ga = (uint64_t)(size_t)&Wt[k0];
            uint32_t la = (uint32_t)(size_t)&lWt[0];
            u32x4 g0;
            g0[0] = 1u;                                     // count=1 (valid user D#)
            g0[1] = la;                                     // lds_addr (bytes)
            g0[2] = (uint32_t)ga;                           // global_addr[31:0]
            g0[3] = (uint32_t)((ga >> 32) & 0x1FFFFFFu) | (2u << 30);  // addr[56:32] | type=2
            i32x8 g1;
            g1[0] = 0x10000;                                // data_size=1 (2 bytes)
            g1[1] = (int)(((unsigned)K & 0xFFFFu) << 16);   // tensor_dim0[15:0]
            g1[2] = (int)(128u << 16);                      // tensor_dim1 = 128
            g1[3] = (int)(32u << 16);                       // tile_dim0 = 32
            g1[4] = 128;                                    // tile_dim1 = 128
            g1[5] = K;                                      // tensor_dim0_stride
            g1[6] = 0; g1[7] = 0;
            i32x4 z4 = {0, 0, 0, 0};
#  if __clang_major__ >= 23
            i32x8 z8 = {0, 0, 0, 0, 0, 0, 0, 0};
            __builtin_amdgcn_tensor_load_to_lds(g0, g1, z4, z4, z8, 0);
#  else
            __builtin_amdgcn_tensor_load_to_lds(g0, g1, z4, z4, 0);
#  endif
            __builtin_amdgcn_s_wait_tensorcnt((short)0);
        }
#else
        // fallback: vector-copy pre-transposed f16 tile (512 x 16B segments / 2 per thread)
#pragma unroll
        for (int it = 0; it < 2; ++it) {
            int seg = tid + it * 256;      // 0..511
            int c = seg >> 2, s8 = (seg & 3) * 8;
            *(v8h*)&lWt[c * 32 + s8] = *(const v8h*)&Wt[(size_t)c * K + k0 + s8];
        }
#endif
        __syncthreads();

        // B fragment: 16 contiguous halves -> 2x ds_load_b128
        v8h b0 = *(const v8h*)&lWt[ncol * 32 + g * 16];
        v8h b1 = *(const v8h*)&lWt[ncol * 32 + g * 16 + 8];
        v16h bf = __builtin_shufflevector(b0, b1, 0,1,2,3,4,5,6,7,8,9,10,11,12,13,14,15);
#pragma unroll
        for (int t = 0; t < 4; ++t) {
            int mrow = t * 16 + mr;
            v8h a0 = *(const v8h*)&lA[mrow][g * 16];
            v8h a1 = *(const v8h*)&lA[mrow][g * 16 + 8];
            v16h af = __builtin_shufflevector(a0, a1, 0,1,2,3,4,5,6,7,8,9,10,11,12,13,14,15);
            acc[t] = __builtin_amdgcn_wmma_f32_16x16x32_f16(false, af, false, bf,
                                                            (short)0, acc[t], false, false);
        }
        __syncthreads();
    }
    float bval = bias[ncol];
#pragma unroll
    for (int t = 0; t < 4; ++t) {
#pragma unroll
        for (int r = 0; r < 8; ++r) {
            int gr = row0 + t * 16 + r + 8 * g;
            float v = acc[t][r] + bval;
            if (flags & 1) v = fmaxf(v, 0.f);
            if (flags & 2) v += resid[(size_t)gr * 128 + ncol];
            C[(size_t)gr * 128 + ncol] = v;
        }
    }
}

// ============ layer-1 encoders ============
__global__ void enc_edge_l1(const float* __restrict__ state, const float* __restrict__ control,
                            const float* __restrict__ W1, const float* __restrict__ b1,
                            float* __restrict__ h)
{
    int i = blockIdx.x, j = threadIdx.x;
    int t = edge_type(i);
    float x;
    if (i < 2048)      x = state[i];
    else if (i < 3072) x = 0.f;
    else               x = control[i - 3072];
    float v = x * W1[t * 128 + j] + b1[t * 128 + j];
    h[(size_t)i * 128 + j] = fmaxf(v, 0.f);
}

__global__ void enc_node_l1(const float* __restrict__ state,
                            const float* __restrict__ W1, const float* __restrict__ b1,
                            float* __restrict__ h)
{
    int i = blockIdx.x, j = threadIdx.x;
    float x = (i >= 1 && i < N_GNODES) ? state[NCd + NLd + i - 1] : 0.f;
    float v = x * W1[j] + b1[j];
    h[(size_t)i * 128 + j] = fmaxf(v, 0.f);
}

// ============ LayerNorm over 128 ============
__global__ void ln_kernel(const float* __restrict__ X, const float* __restrict__ S,
                          const float* __restrict__ B, float* __restrict__ Y, int typed)
{
    __shared__ float red[128];
    int row = blockIdx.x, j = threadIdx.x;
    float v = X[(size_t)row * 128 + j];
    red[j] = v; __syncthreads();
    for (int s = 64; s > 0; s >>= 1) { if (j < s) red[j] += red[j + s]; __syncthreads(); }
    float mean = red[0] * (1.f / 128.f);
    __syncthreads();
    float d = v - mean;
    red[j] = d * d; __syncthreads();
    for (int s = 64; s > 0; s >>= 1) { if (j < s) red[j] += red[j + s]; __syncthreads(); }
    float var = red[0] * (1.f / 128.f);
    int t = typed ? edge_type(row) : 0;
    Y[(size_t)row * 128 + j] = d * rsqrtf(var + 1e-6f) * S[t * 128 + j] + B[t * 128 + j];
}

// ============ message-passing plumbing ============
__global__ void gather_m(const float* __restrict__ e, const float* __restrict__ n,
                         const int* __restrict__ snd, const int* __restrict__ rcv,
                         float* __restrict__ m)
{
    int i = blockIdx.x, j = threadIdx.x;
    m[(size_t)i * 384 + j]       = e[(size_t)i * 128 + j];
    m[(size_t)i * 384 + 128 + j] = n[(size_t)snd[i] * 128 + j];
    m[(size_t)i * 384 + 256 + j] = n[(size_t)rcv[i] * 128 + j];
}

__global__ void fill0(float* __restrict__ p, int nelem)
{
    int i = blockIdx.x * 256 + threadIdx.x;
    if (i < nelem) p[i] = 0.f;
}

__global__ void scatter_e(const float* __restrict__ e, const int* __restrict__ rcv,
                          float* __restrict__ agg)
{
    int i = blockIdx.x, j = threadIdx.x;
    atomicAdd(&agg[(size_t)rcv[i] * 128 + j], e[(size_t)i * 128 + j]);
}

__global__ void gather_u(const float* __restrict__ n, const float* __restrict__ agg,
                         float* __restrict__ u)
{
    int i = blockIdx.x, j = threadIdx.x;
    u[(size_t)i * 256 + j]       = n[(size_t)i * 128 + j];
    u[(size_t)i * 256 + 128 + j] = agg[(size_t)i * 128 + j];
}

// ============ decoder output heads (128 -> 1 dot) ============
__global__ void dec_out(const float* __restrict__ H2, const float* __restrict__ W3,
                        const float* __restrict__ b3, float* __restrict__ out, int typed)
{
    __shared__ float red[128];
    int row = blockIdx.x, j = threadIdx.x;
    int t = typed ? edge_type(row) : 0;
    red[j] = H2[(size_t)row * 128 + j] * W3[t * 128 + j];
    __syncthreads();
    for (int s = 64; s > 0; s >>= 1) { if (j < s) red[j] += red[j + s]; __syncthreads(); }
    if (j == 0) out[row] = red[0] + b3[t];
}

// ============ misc assembly ============
__global__ void build_next_y(const float* __restrict__ n_dec, const float* __restrict__ e_dec,
                             float* __restrict__ ny)
{
    int i = blockIdx.x * 256 + threadIdx.x;
    if (i < 4096)      ny[i] = n_dec[1 + i];
    else if (i < NALG) ny[i] = e_dec[3072 + (i - 4096)];
}

__global__ void dot_include(const float* __restrict__ inc, const float* __restrict__ edec,
                            float* __restrict__ out)
{
    __shared__ float red[256];
    float s = 0.f;
    for (int i = threadIdx.x; i < N_EDGES; i += 256) s += inc[i] * edec[i];
    red[threadIdx.x] = s; __syncthreads();
    for (int st = 128; st > 0; st >>= 1) { if (threadIdx.x < st) red[threadIdx.x] += red[threadIdx.x + st]; __syncthreads(); }
    if (threadIdx.x == 0) out[0] = red[0];
}

// ============ matvecs (32 lanes/row, float4 loads, shfl reduce) ============
__device__ __forceinline__ float row_dot(const float* __restrict__ Jr,
                                         const float* __restrict__ v, int K, int lane)
{
    float s = 0.f;
    for (int k = lane * 4; k < K; k += 128) {
        f4 a = *(const f4*)&Jr[k];
        f4 b = *(const f4*)&v[k];
        s += a[0] * b[0] + a[1] * b[1] + a[2] * b[2] + a[3] * b[3];
    }
    for (int off = 16; off; off >>= 1) s += __shfl_xor(s, off, 32);
    return s;
}

__global__ __launch_bounds__(256) void matvec_B(const float* __restrict__ Bm,
                                                const float* __restrict__ ctrl,
                                                float* __restrict__ Bu)
{
    int lane = threadIdx.x & 31;
    int r = blockIdx.x * 8 + (threadIdx.x >> 5);
    if (r >= STATE_DIM) return;
    float s = row_dot(Bm + (size_t)r * N_CTRL, ctrl, N_CTRL, lane);
    if (lane == 0) Bu[r] = s;
}

__global__ __launch_bounds__(256) void cvec_mv(const float* __restrict__ J,
                                               const float* __restrict__ ny,
                                               const float* __restrict__ Bu,
                                               float* __restrict__ cvec)
{
    int lane = threadIdx.x & 31;
    int r = blockIdx.x * 8 + (threadIdx.x >> 5);
    if (r >= NDIFF) return;
    float s = row_dot(J + (size_t)r * STATE_DIM + NDIFF, ny, NALG, lane);
    if (lane == 0) cvec[r] = s + Bu[r];
}

__global__ __launch_bounds__(256) void f_eval(const float* __restrict__ J,
                                              const float* __restrict__ x,
                                              const float* __restrict__ cvec,
                                              const float* __restrict__ r_diag,
                                              float* __restrict__ out)
{
    int lane = threadIdx.x & 31;
    int r = blockIdx.x * 8 + (threadIdx.x >> 5);
    if (r >= NDIFF) return;
    float s = row_dot(J + (size_t)r * STATE_DIM, x, NDIFF, lane);
    if (lane == 0) out[r] = s + cvec[r] - r_diag[r] * x[r];
}

__global__ void ab2_update(float* __restrict__ x, float* __restrict__ fp,
                           const float* __restrict__ fc)
{
    int i = blockIdx.x * 256 + threadIdx.x;
    if (i < NDIFF) {
        x[i] += DT_C * (1.5f * fc[i] - 0.5f * fp[i]);
        fp[i] = fc[i];
    }
}

__global__ void copy_x0(const float* __restrict__ state, float* __restrict__ x)
{
    int i = blockIdx.x * 256 + threadIdx.x;
    if (i < NDIFF) x[i] = state[i];
}

__global__ void final_state(const float* __restrict__ x, const float* __restrict__ ny,
                            float* __restrict__ out)
{
    int i = blockIdx.x * 256 + threadIdx.x;
    if (i < NDIFF)          out[i] = x[i];
    else if (i < STATE_DIM) out[i] = ny[i - NDIFF];
}

__global__ __launch_bounds__(256) void resid_mv(const float* __restrict__ J,
                                                const float* __restrict__ r_diag,
                                                const float* __restrict__ Bu,
                                                float* __restrict__ out)
{
    int lane = threadIdx.x & 31;
    int r = blockIdx.x * 8 + (threadIdx.x >> 5);
    if (r >= NALG) return;
    int row = NDIFF + r;
    float s = row_dot(J + (size_t)row * STATE_DIM, out, STATE_DIM, lane);
    if (lane == 0) out[STATE_DIM + 1 + r] = fabsf(s - r_diag[row] * out[row] + Bu[row]);
}

// =============================== host ===============================
extern "C" void kernel_launch(void* const* d_in, const int* in_sizes, int n_in,
                              void* d_out, int out_size, void* d_ws, size_t ws_size,
                              hipStream_t stream)
{
    const float* state   = (const float*)d_in[0];
    const float* control = (const float*)d_in[1];
    const float* J       = (const float*)d_in[2];
    const float* Bm      = (const float*)d_in[3];
    const float* r_diag  = (const float*)d_in[4];
    const float* inc     = (const float*)d_in[5];
    const float* eeW1 = (const float*)d_in[6],  *eeb1 = (const float*)d_in[7];
    const float* eeW2 = (const float*)d_in[8],  *eeb2 = (const float*)d_in[9];
    const float* eels = (const float*)d_in[10], *eelb = (const float*)d_in[11];
    const float* enW1 = (const float*)d_in[12], *enb1 = (const float*)d_in[13];
    const float* enW2 = (const float*)d_in[14], *enb2 = (const float*)d_in[15];
    const float* enls = (const float*)d_in[16], *enlb = (const float*)d_in[17];
    const float* meW1 = (const float*)d_in[18], *meb1 = (const float*)d_in[19];
    const float* meW2 = (const float*)d_in[20], *meb2 = (const float*)d_in[21];
    const float* mnW1 = (const float*)d_in[22], *mnb1 = (const float*)d_in[23];
    const float* mnW2 = (const float*)d_in[24], *mnb2 = (const float*)d_in[25];
    const float* deW1 = (const float*)d_in[26], *deb1 = (const float*)d_in[27];
    const float* deW2 = (const float*)d_in[28], *deb2 = (const float*)d_in[29];
    const float* deW3 = (const float*)d_in[30], *deb3 = (const float*)d_in[31];
    const float* dnW1 = (const float*)d_in[32], *dnb1 = (const float*)d_in[33];
    const float* dnW2 = (const float*)d_in[34], *dnb2 = (const float*)d_in[35];
    const float* dnW3 = (const float*)d_in[36], *dnb3 = (const float*)d_in[37];
    const int* senders   = (const int*)d_in[38];
    const int* receivers = (const int*)d_in[39];
    float* out = (float*)d_out;

    // workspace carve-up (float units; all offsets stay 16B-aligned)
    float* ws = (float*)d_ws;
    size_t o = 0;
    auto alloc = [&](size_t n) { float* p = ws + o; o += n; return p; };
    float* h_e  = alloc((size_t)N_EDGES * 128);
    float* e    = alloc((size_t)N_EDGES * 128);
    float* t_e  = alloc((size_t)N_EDGES * 128);
    float* t_e2 = alloc((size_t)N_EDGES * 128);
    float* nbuf = alloc((size_t)N_GPAD * 128);
    float* h_n  = alloc((size_t)N_GPAD * 128);
    float* t_n  = alloc((size_t)N_GPAD * 128);
    float* agg  = alloc((size_t)N_GPAD * 128);
    float* mbuf = alloc((size_t)N_EDGES * 384);
    float* ubuf = alloc((size_t)N_GPAD * 256);
    float* e_dec = alloc(N_EDGES);
    float* n_dec = alloc(N_GPAD);
    float* ny    = alloc(NALG);
    float* Bu    = alloc(STATE_DIM);
    float* cvec  = alloc(NDIFF);
    float* x     = alloc(NDIFF);
    float* fc    = alloc(NDIFF);
    float* fp    = alloc(NDIFF);
    // f16 transposed weights (409600 halves = 204800 floats)
    _Float16* wt0 = (_Float16*)alloc(204800);
    _Float16* eeW2t = wt0;
    _Float16* enW2t = eeW2t + 5 * 16384;
    _Float16* meW1t = enW2t + 16384;
    _Float16* meW2t = meW1t + 49152;
    _Float16* mnW1t = meW2t + 16384;
    _Float16* mnW2t = mnW1t + 32768;
    _Float16* deW1t = mnW2t + 16384;
    _Float16* deW2t = deW1t + 5 * 16384;
    _Float16* dnW1t = deW2t + 5 * 16384;
    _Float16* dnW2t = dnW1t + 16384;
    (void)ws_size; (void)n_in; (void)in_sizes; (void)out_size;

    const int toff[5] = {0, 1024, 2048, 3072, 3584};
    const int tsz[5]  = {1024, 1024, 1024, 512, 512};

    // ---- transpose+convert all GEMM weights to f16 (tiny; once per call) ----
    for (int t = 0; t < 5; ++t) {
        w_transpose<<<64, 256, 0, stream>>>(eeW2 + (size_t)t * 16384, eeW2t + (size_t)t * 16384, 128);
        w_transpose<<<64, 256, 0, stream>>>(deW1 + (size_t)t * 16384, deW1t + (size_t)t * 16384, 128);
        w_transpose<<<64, 256, 0, stream>>>(deW2 + (size_t)t * 16384, deW2t + (size_t)t * 16384, 128);
    }
    w_transpose<<<64,  256, 0, stream>>>(enW2, enW2t, 128);
    w_transpose<<<192, 256, 0, stream>>>(meW1, meW1t, 384);
    w_transpose<<<64,  256, 0, stream>>>(meW2, meW2t, 128);
    w_transpose<<<128, 256, 0, stream>>>(mnW1, mnW1t, 256);
    w_transpose<<<64,  256, 0, stream>>>(mnW2, mnW2t, 128);
    w_transpose<<<64,  256, 0, stream>>>(dnW1, dnW1t, 128);
    w_transpose<<<64,  256, 0, stream>>>(dnW2, dnW2t, 128);

    // ---- encoder ----
    enc_edge_l1<<<N_EDGES, 128, 0, stream>>>(state, control, eeW1, eeb1, h_e);
    for (int t = 0; t < 5; ++t)
        gemm128_wmma<<<tsz[t] / 64, 256, 0, stream>>>(h_e + (size_t)toff[t] * 128, 128,
            eeW2t + (size_t)t * 16384, eeb2 + t * 128, nullptr,
            t_e + (size_t)toff[t] * 128, 128, 0);
    ln_kernel<<<N_EDGES, 128, 0, stream>>>(t_e, eels, eelb, e, 1);

    enc_node_l1<<<N_GPAD, 128, 0, stream>>>(state, enW1, enb1, h_n);
    gemm128_wmma<<<N_GPAD / 64, 256, 0, stream>>>(h_n, 128, enW2t, enb2, nullptr, t_n, 128, 0);
    ln_kernel<<<N_GPAD, 128, 0, stream>>>(t_n, enls, enlb, nbuf, 0);

    // ---- 3 message-passing rounds ----
    for (int mp = 0; mp < 3; ++mp) {
        gather_m<<<N_EDGES, 128, 0, stream>>>(e, nbuf, senders, receivers, mbuf);
        gemm128_wmma<<<N_EDGES / 64, 256, 0, stream>>>(mbuf, 384, meW1t, meb1, nullptr, t_e, 384, 1);
        gemm128_wmma<<<N_EDGES / 64, 256, 0, stream>>>(t_e, 128, meW2t, meb2, e, e, 128, 2);
        fill0<<<(N_GPAD * 128 + 255) / 256, 256, 0, stream>>>(agg, N_GPAD * 128);
        scatter_e<<<N_EDGES, 128, 0, stream>>>(e, receivers, agg);
        gather_u<<<N_GPAD, 128, 0, stream>>>(nbuf, agg, ubuf);
        gemm128_wmma<<<N_GPAD / 64, 256, 0, stream>>>(ubuf, 256, mnW1t, mnb1, nullptr, t_n, 256, 1);
        gemm128_wmma<<<N_GPAD / 64, 256, 0, stream>>>(t_n, 128, mnW2t, mnb2, nbuf, nbuf, 128, 2);
    }

    // ---- decoders ----
    for (int t = 0; t < 5; ++t) {
        gemm128_wmma<<<tsz[t] / 64, 256, 0, stream>>>(e + (size_t)toff[t] * 128, 128,
            deW1t + (size_t)t * 16384, deb1 + t * 128, nullptr,
            t_e + (size_t)toff[t] * 128, 128, 1);
        gemm128_wmma<<<tsz[t] / 64, 256, 0, stream>>>(t_e + (size_t)toff[t] * 128, 128,
            deW2t + (size_t)t * 16384, deb2 + t * 128, nullptr,
            t_e2 + (size_t)toff[t] * 128, 128, 1);
    }
    dec_out<<<N_EDGES, 128, 0, stream>>>(t_e2, deW3, deb3, e_dec, 1);

    gemm128_wmma<<<N_GPAD / 64, 256, 0, stream>>>(nbuf, 128, dnW1t, dnb1, nullptr, h_n, 128, 1);
    gemm128_wmma<<<N_GPAD / 64, 256, 0, stream>>>(h_n, 128, dnW2t, dnb2, nullptr, t_n, 128, 1);
    dec_out<<<N_GNODES, 128, 0, stream>>>(t_n, dnW3, dnb3, n_dec, 0);

    // ---- outputs of GNN: next_y, H ----
    build_next_y<<<(NALG + 255) / 256, 256, 0, stream>>>(n_dec, e_dec, ny);
    dot_include<<<1, 256, 0, stream>>>(inc, e_dec, out + STATE_DIM);

    // ---- dynamics: Bu, constant part, AB2 integration ----
    matvec_B<<<STATE_DIM / 8, 256, 0, stream>>>(Bm, control, Bu);
    cvec_mv<<<NDIFF / 8, 256, 0, stream>>>(J, ny, Bu, cvec);
    copy_x0<<<NDIFF / 256, 256, 0, stream>>>(state, x);
    f_eval<<<NDIFF / 8, 256, 0, stream>>>(J, x, cvec, r_diag, fp);   // fp = f(x0)
    for (int s = 0; s < 10; ++s) {
        f_eval<<<NDIFF / 8, 256, 0, stream>>>(J, x, cvec, r_diag, fc);
        ab2_update<<<NDIFF / 256, 256, 0, stream>>>(x, fp, fc);
    }

    // ---- final state + residuals ----
    final_state<<<(STATE_DIM + 255) / 256, 256, 0, stream>>>(x, ny, out);
    resid_mv<<<NALG / 8, 256, 0, stream>>>(J, r_diag, Bu, out);
}
